// MaatConstrainedAttention_23940147708331
// MI455X (gfx1250) — compile-verified, matrix operand order checked by reading, not run on hardware
//
#include <hip/hip_runtime.h>
#include <hip/hip_bf16.h>

// ---------------------------------------------------------------------------
// Maat constrained attention, CDNA5/gfx1250.
//   S=2048 seq, D=2048 hidden, H=16 heads, hd=128.
// Pipeline:
//   1) cast hidden + 4 weights fp32 -> fp16 (workspace)
//   2) WMMA GEMM x@W^T: 4-wave workgroups, 64(M)x64(N) tile; the shared B
//      strip is double-buffered in LDS via global_load_async_to_lds_b128
//      (ASYNCcnt DMA; falls back to ds_store staging if builtin absent),
//      A fragments direct from global with 1-deep preload.
//   3) fused attention: per (head, 16-query strip) one wave:
//        scores strip (16x2048) via v_wmma_f32_16x16x32_f16 -> f16 LDS (64KB),
//        row softmax keeping only softmax-relevant gate terms
//        (justice scale folded into QK scale, order key-bias added inline;
//         truth/balance/harmony/justice-mean are per-row constants -> cancel),
//        P@V via WMMA (P from LDS, V^T streamed from L2) -> f16 context
//   4) WMMA GEMM ctx@Wo^T + bo -> fp32 d_out
// ---------------------------------------------------------------------------

#define S_LEN 2048
#define DMODEL 2048
#define NHEADS 16
#define HDIM 128

typedef _Float16 f16;
typedef __attribute__((ext_vector_type(16))) _Float16 v16h;
typedef __attribute__((ext_vector_type(8)))  _Float16 v8h;
typedef __attribute__((ext_vector_type(8)))  float    v8f;

#if defined(__has_builtin)
#if __has_builtin(__builtin_amdgcn_global_load_async_to_lds_b128) && \
    __has_builtin(__builtin_amdgcn_s_wait_asynccnt)
#define HAVE_ASYNC_LDS 1
#endif
#endif

#if defined(HAVE_ASYNC_LDS)
// builtin prototype: (v4i addrspace(1)* src, v4i addrspace(3)* dst, Imm, Imm)
typedef int vi4 __attribute__((vector_size(16)));
typedef __attribute__((address_space(1))) vi4* gvi4_p;
typedef __attribute__((address_space(3))) vi4* lvi4_p;
#endif

__device__ __forceinline__ v8f zero8() {
    v8f z;
#pragma unroll
    for (int i = 0; i < 8; ++i) z[i] = 0.0f;
    return z;
}

// Load one 16x32 (A, M-major) or 32x16 (B, N-major) f16 WMMA fragment.
// Row/col index = lane&15 (stride `ld` f16 elems); K pattern per CDNA5 ISA:
// lanes 0-15 hold K={0..7,16..23}, lanes 16-31 hold K={8..15,24..31}.
__device__ __forceinline__ v16h load_frag(const f16* __restrict__ base, int ld, int lane) {
    const f16* p = base + (lane & 15) * ld + ((lane >> 4) << 3);
    union { v16h v; v8h h[2]; } u;
    u.h[0] = *(const v8h*)(p);       // K = kbase + 0..7
    u.h[1] = *(const v8h*)(p + 16);  // K = kbase + 16..23
    return u.v;
}

__device__ __forceinline__ v8f wmma_f16(v16h a, v16h b, v8f c) {
    return __builtin_amdgcn_wmma_f32_16x16x32_f16(
        /*neg_a=*/false, a, /*neg_b=*/false, b,
        /*c_mod=*/(short)0, c, /*reuse_a=*/false, /*reuse_b=*/false);
}

// ---------------------------------------------------------------------------
// fp32 -> fp16 cast, 4 elems/thread (n multiple of 4)
// ---------------------------------------------------------------------------
__global__ void cast_f32_f16(const float* __restrict__ src, f16* __restrict__ dst, int n) {
    int i = (blockIdx.x * blockDim.x + threadIdx.x) * 4;
    if (i + 3 < n) {
        float4 v = *(const float4*)(src + i);
        dst[i + 0] = (f16)v.x; dst[i + 1] = (f16)v.y;
        dst[i + 2] = (f16)v.z; dst[i + 3] = (f16)v.w;
    }
}

// ---------------------------------------------------------------------------
// out[m,n] = sum_k X[m,k] * W[n,k] + bias[n]   (X: SxD f16, W: DxD f16)
// 4 waves per workgroup; workgroup tile 64(M) x 64(N); wave w owns rows
// m0+16w..m0+16w+15. The 64x32 B K-slab is shared via LDS, double-buffered:
// staged by async LDS-DMA (global_load_async_to_lds_b128, ASYNCcnt) while the
// current slab feeds 4 WMMAs per wave; one barrier per K-step.
// mode 0: f16 out, row-major [S][D]     (Q, K projections)
// mode 1: f16 out, transposed [n][S]    (V projection -> V^T for P@V B-loads)
// mode 2: f32 out, row-major [S][D]     (output projection -> d_out)
// ---------------------------------------------------------------------------
__global__ void __launch_bounds__(128, 1)
gemm_xwT(const f16* __restrict__ X, const f16* __restrict__ W,
         const float* __restrict__ bias, void* __restrict__ out, int mode) {
    __shared__ alignas(16) f16 Bs[2][64][32];   // 2 x 4 KB B slab

    const int tid  = threadIdx.x;
    const int lane = tid & 31;
    const int wave = tid >> 5;
    const int m0 = blockIdx.x * 64;
    const int n0 = blockIdx.y * 64;

    // stage 64 rows x 32 cols of W (K-slab at k0) into Bs[buf]:
    // 256 16-byte chunks, 2 per thread, DMA'd straight into LDS.
    auto stage = [&](int buf, int k0) {
#pragma unroll
        for (int i = 0; i < 2; ++i) {
            const int c = tid + i * 128;
            const int row = c >> 2, part = c & 3;
            const f16* src = W + (n0 + row) * DMODEL + k0 + part * 8;
            f16* dst = &Bs[buf][row][part * 8];
#if defined(HAVE_ASYNC_LDS)
            __builtin_amdgcn_global_load_async_to_lds_b128(
                (gvi4_p)(f16*)src, (lvi4_p)dst, 0, 0);
#else
            *(v8h*)dst = *(const v8h*)src;
#endif
        }
    };

    v8f acc[4];
#pragma unroll
    for (int t = 0; t < 4; ++t) acc[t] = zero8();

    const f16* Arow = X + (m0 + wave * 16) * DMODEL;

    stage(0, 0);
    v16h a0 = load_frag(Arow, DMODEL, lane);
#if defined(HAVE_ASYNC_LDS)
    __builtin_amdgcn_s_wait_asynccnt(0);
#endif
    __syncthreads();

    for (int k0 = 0; k0 < DMODEL; k0 += 32) {
        const int cur = (k0 >> 5) & 1;
        v16h a1;
        if (k0 + 32 < DMODEL) {
            stage(cur ^ 1, k0 + 32);                     // DMA next slab
            a1 = load_frag(Arow + k0 + 32, DMODEL, lane); // preload next A
            __builtin_prefetch(Arow + (lane & 15) * DMODEL + k0 + 288, 0, 1);
        }
        v16h b[4];
#pragma unroll
        for (int t = 0; t < 4; ++t)
            b[t] = load_frag(&Bs[cur][t * 16][0], 32, lane);   // ds_load_b128
#pragma unroll
        for (int t = 0; t < 4; ++t)
            acc[t] = wmma_f16(a0, b[t], acc[t]);
        if (k0 + 32 < DMODEL) a0 = a1;
#if defined(HAVE_ASYNC_LDS)
        __builtin_amdgcn_s_wait_asynccnt(0);
#endif
        __syncthreads();
    }

    // C/D layout: lanes 0-15: (M=r, N=lane); lanes 16-31: (M=r+8, N=lane-16)
    const int mrow = (lane >> 4) << 3;
    const int ncol = lane & 15;
#pragma unroll
    for (int t = 0; t < 4; ++t) {
        const int n = n0 + t * 16 + ncol;
        const float bn = bias[n];
#pragma unroll
        for (int r = 0; r < 8; ++r) {
            const int m = m0 + wave * 16 + r + mrow;
            const float v = acc[t][r] + bn;
            if (mode == 0)      ((f16*)out)[m * DMODEL + n] = (f16)v;
            else if (mode == 1) ((f16*)out)[n * S_LEN + m]  = (f16)v;   // V^T
            else                ((float*)out)[m * DMODEL + n] = v;
        }
    }
}

// ---------------------------------------------------------------------------
// Fused attention: one wave per (16-query strip, head).
// Only softmax-relevant gate terms kept (see header comment).
// ---------------------------------------------------------------------------
__global__ void __launch_bounds__(32, 1)
maat_attention(const f16* __restrict__ Q, const f16* __restrict__ K,
               const f16* __restrict__ Vt, f16* __restrict__ Ctx,
               const float* __restrict__ justice_gate,
               const float* __restrict__ order_gate) {
    __shared__ alignas(16) f16 Ps[16 * S_LEN];   // 64 KB score / prob strip

    const int lane = threadIdx.x;
    const int q0 = blockIdx.x * 16;
    const int h  = blockIdx.y;
    const int hoff = h * HDIM;

    const float sj = 1.0f / (1.0f + __expf(-justice_gate[0]));
    const float so = 1.0f / (1.0f + __expf(-order_gate[0]));
    const float alpha = (1.0f - 0.1f * sj) * 0.088388347648318447f; // (1-0.1σj)/sqrt(128)
    const float beta  = 0.05f * so / (float)S_LEN;                  // order key bias

    // Q fragments for this strip: 16 x 128, reused over all 128 key tiles
    v16h aQ[4];
#pragma unroll
    for (int d = 0; d < 4; ++d)
        aQ[d] = load_frag(Q + q0 * DMODEL + hoff + d * 32, DMODEL, lane);

    const int mrow = (lane >> 4) << 3;
    const int ncol = lane & 15;

    // compute one 16x16 score tile from 4 K-fragments, store f16 into LDS
    auto score_tile = [&](int kt, const v16h* bK) {
        v8f s = zero8();
#pragma unroll
        for (int d = 0; d < 4; ++d)
            s = wmma_f16(aQ[d], bK[d], s);
        const int n = kt * 16 + ncol;
        const float kb = beta * (float)n;
#pragma unroll
        for (int r = 0; r < 8; ++r)
            Ps[(r + mrow) * S_LEN + n] = (f16)(s[r] * alpha + kb);
    };

    // ---- scores strip into LDS (f16), 2x-unrolled ping-pong K-tile loads ----
    {
        const int NT = S_LEN / 16;               // 128 key tiles
        v16h b0[4], b1[4];
#pragma unroll
        for (int d = 0; d < 4; ++d)
            b0[d] = load_frag(K + hoff + d * 32, DMODEL, lane);
        int kt = 0;
        for (; kt < NT - 2; kt += 2) {
#pragma unroll
            for (int d = 0; d < 4; ++d)
                b1[d] = load_frag(K + (kt + 1) * 16 * DMODEL + hoff + d * 32, DMODEL, lane);
            score_tile(kt, b0);
#pragma unroll
            for (int d = 0; d < 4; ++d)
                b0[d] = load_frag(K + (kt + 2) * 16 * DMODEL + hoff + d * 32, DMODEL, lane);
            score_tile(kt + 1, b1);
        }
#pragma unroll
        for (int d = 0; d < 4; ++d)
            b1[d] = load_frag(K + (kt + 1) * 16 * DMODEL + hoff + d * 32, DMODEL, lane);
        score_tile(kt, b0);
        score_tile(kt + 1, b1);
    }
    __syncthreads();

    // ---- row softmax: 2 lanes per row, each scans half the row ----
    {
        const int row = lane & 15;
        const int half = lane >> 4;
        f16* prow = Ps + row * S_LEN + half * (S_LEN / 2);
        float m = -1e30f;
        for (int i = 0; i < S_LEN / 2; ++i) m = fmaxf(m, (float)prow[i]);
        m = fmaxf(m, __shfl_xor(m, 16, 32));
        float sum = 0.0f;
        for (int i = 0; i < S_LEN / 2; ++i) sum += __expf((float)prow[i] - m);
        sum += __shfl_xor(sum, 16, 32);
        const float inv = 1.0f / sum;
        for (int i = 0; i < S_LEN / 2; ++i)
            prow[i] = (f16)(__expf((float)prow[i] - m) * inv);
    }
    __syncthreads();

    // ---- context = P @ V : A from LDS, B = V^T rows (L2-resident).
    //      All 8 V fragments loaded as one clause before the 8 WMMAs so the
    //      waits form a descending loadcnt ladder instead of 8x wait-0. ----
    v8f o[8];
#pragma unroll
    for (int t = 0; t < 8; ++t) o[t] = zero8();

    for (int kk = 0; kk < S_LEN; kk += 32) {
        v16h a = load_frag(Ps + kk, S_LEN, lane);
        v16h bV[8];
#pragma unroll
        for (int nt = 0; nt < 8; ++nt)
            bV[nt] = load_frag(Vt + (hoff + nt * 16) * S_LEN + kk, S_LEN, lane);
#pragma unroll
        for (int nt = 0; nt < 8; ++nt)
            o[nt] = wmma_f16(a, bV[nt], o[nt]);
    }

#pragma unroll
    for (int nt = 0; nt < 8; ++nt) {
        const int d = hoff + nt * 16 + ncol;
#pragma unroll
        for (int r = 0; r < 8; ++r)
            Ctx[(q0 + r + mrow) * DMODEL + d] = (f16)o[nt][r];
    }
}

// ---------------------------------------------------------------------------
extern "C" void kernel_launch(void* const* d_in, const int* in_sizes, int n_in,
                              void* d_out, int out_size, void* d_ws, size_t ws_size,
                              hipStream_t stream) {
    const float* hs = (const float*)d_in[0];
    const float* Wq = (const float*)d_in[1];  const float* bq = (const float*)d_in[2];
    const float* Wk = (const float*)d_in[3];  const float* bk = (const float*)d_in[4];
    const float* Wv = (const float*)d_in[5];  const float* bv = (const float*)d_in[6];
    const float* Wo = (const float*)d_in[7];  const float* bo = (const float*)d_in[8];
    const float* og = (const float*)d_in[11];
    const float* jg = (const float*)d_in[12];

    const int NE = S_LEN * DMODEL;            // 4M elements per matrix
    f16* ws   = (f16*)d_ws;                   // 72 MB of f16 scratch total
    f16* X16  = ws + (size_t)NE * 0;
    f16* Wq16 = ws + (size_t)NE * 1;
    f16* Wk16 = ws + (size_t)NE * 2;
    f16* Wv16 = ws + (size_t)NE * 3;
    f16* Wo16 = ws + (size_t)NE * 4;
    f16* Q16  = ws + (size_t)NE * 5;
    f16* K16  = ws + (size_t)NE * 6;
    f16* Vt16 = ws + (size_t)NE * 7;          // [channel][seq]
    f16* Ctx  = ws + (size_t)NE * 8;

    const int cthreads = 256;
    const int cblocks  = NE / (cthreads * 4);
    cast_f32_f16<<<cblocks, cthreads, 0, stream>>>(hs, X16, NE);
    cast_f32_f16<<<cblocks, cthreads, 0, stream>>>(Wq, Wq16, NE);
    cast_f32_f16<<<cblocks, cthreads, 0, stream>>>(Wk, Wk16, NE);
    cast_f32_f16<<<cblocks, cthreads, 0, stream>>>(Wv, Wv16, NE);
    cast_f32_f16<<<cblocks, cthreads, 0, stream>>>(Wo, Wo16, NE);

    dim3 gg(S_LEN / 64, DMODEL / 64);         // 32 x 32 workgroups, 64x64 tiles
    gemm_xwT<<<gg, 128, 0, stream>>>(X16, Wq16, bq, Q16, 0);
    gemm_xwT<<<gg, 128, 0, stream>>>(X16, Wk16, bk, K16, 0);
    gemm_xwT<<<gg, 128, 0, stream>>>(X16, Wv16, bv, Vt16, 1);

    maat_attention<<<dim3(S_LEN / 16, NHEADS), 32, 0, stream>>>(Q16, K16, Vt16, Ctx, jg, og);

    gemm_xwT<<<gg, 128, 0, stream>>>(Ctx, Wo16, bo, d_out, 2);
}
